// GraphInformer_6073083757203
// MI455X (gfx1250) — compile-verified, matrix operand order checked by reading, not run on hardware
//
#include <hip/hip_runtime.h>
#include <hip/hip_bf16.h>
#include <math.h>

// ---------------------------------------------------------------------------
// Types for CDNA5 WMMA
// ---------------------------------------------------------------------------
typedef __attribute__((ext_vector_type(16))) __bf16 v16bf;
typedef __attribute__((ext_vector_type(8)))  __bf16 v8bf;
typedef __attribute__((ext_vector_type(8)))  float  v8f;

#define HEADS 8
#define KS    32
#define KRS   16
#define VS    32
#define TOT   112      // 2*KS + VS + KRS
#define NB    8        // batch
#define NN    192      // nodes
#define HID   256
#define NF    32
#define RF    8
#define RS    16
#define IM    1024
#define NL    3
#define SCALE_F 0.14433756729740643f   // 1/sqrt(48)

// float -> bf16, round to nearest even (bit manipulation; no __bf16 arith)
__device__ __forceinline__ __bf16 f2bf(float f) {
  union { float f; unsigned u; } a; a.f = f;
  unsigned r = a.u + 0x7FFFu + ((a.u >> 16) & 1u);
  union { unsigned short s; __bf16 b; } c;
  c.s = (unsigned short)(r >> 16);
  return c.b;
}

// Load a 16x32 bf16 fragment (A layout, or B layout from a transposed tile).
// ISA 7.12.2: lane l<16 holds row (l&15), K-chunks [0..7] and [16..23];
// lane l>=16 holds the same row, K-chunks [8..15] and [24..31].
__device__ __forceinline__ v16bf load_frag(const __bf16* base, int row_stride) {
  int lane = threadIdx.x & 31;
  int row  = lane & 15;
  int ko   = (lane >> 4) << 3;            // 0 or 8
  const __bf16* p = base + row * row_stride;
  v8bf lo = *(const v8bf*)(p + ko);
  v8bf hi = *(const v8bf*)(p + 16 + ko);
  return __builtin_shufflevector(lo, hi, 0,1,2,3,4,5,6,7,8,9,10,11,12,13,14,15);
}

__device__ __forceinline__ v8f wmma_bf16(v16bf a, v16bf b, v8f c) {
  return __builtin_amdgcn_wmma_f32_16x16x32_bf16(false, a, false, b,
                                                 (short)0, c, false, false);
}

// ---------------------------------------------------------------------------
// Pack f32 weight [K,N] into bf16 WMMA-B fragment tiles:
//   tile (kt, nt) is 512 elems at ((kt*Ntiles)+nt)*512, tile[nl*32+kk] =
//   W[kt*32+kk][nt*16+nl], zero-padded.  B fragments then load straight from
//   global memory (2 x b128 per lane), no LDS transpose in the GEMM hot loop.
// ---------------------------------------------------------------------------
__global__ void pack_bf16_kernel(const float* __restrict__ src,
                                 __bf16* __restrict__ dst,
                                 int K, int N, int Ntiles, long total) {
  long idx = (long)blockIdx.x * blockDim.x + threadIdx.x;
  long stride = (long)gridDim.x * blockDim.x;
  for (; idx < total; idx += stride) {
    int  within = (int)(idx & 511);
    long tile   = idx >> 9;
    int nl = within >> 5, kk = within & 31;
    long kt = tile / Ntiles;
    int  nt = (int)(tile - kt * Ntiles);
    long k = kt * 32 + kk;
    int  n = nt * 16 + nl;
    float v = (k < K && n < N) ? src[k * (long)N + n] : 0.0f;
    dst[idx] = f2bf(v);
  }
}

// ---------------------------------------------------------------------------
// N-major WMMA GEMM: block = 4 waves sharing one 16-row A tile via LDS;
// each wave owns 4 consecutive N-tiles (4 accumulators, A frag reused 4x).
// Block output tile: 16 x 256.
// REQUIRES: M % 16 == 0 and K % 32 == 0 (true for all call sites) -> the
// staging and epilogue are completely branch-free in M/K.
// act: 0=none, 1=tanh, 2=relu
// ---------------------------------------------------------------------------
__global__ __launch_bounds__(128)
void gemm_nmajor_kernel(const float* __restrict__ A, int lda,
                        const __bf16* __restrict__ Bp, int Ntiles,
                        const float* __restrict__ bias,
                        const float* __restrict__ resid, int ldr,
                        float* __restrict__ C, int ldc,
                        int M, int Ntot, int K, int act) {
  __shared__ __align__(16) __bf16 As[512];

  const int m0   = blockIdx.x * 16;
  const int tid  = threadIdx.x;
  const int wave = tid >> 5;
  const int lane = tid & 31;
  const int nt0  = blockIdx.y * 16 + wave * 4;
  const int Ktiles = K >> 5;
  // A staging coords: thread -> one float4 (row tid/8, col (tid%8)*4)
  const int ar = tid >> 3;
  const int ac = (tid & 7) << 2;
  // clamp tile ids so every lane's B loads stay in-bounds (tile 0 always valid)
  int bt[4];
#pragma unroll
  for (int q = 0; q < 4; ++q) { int t = nt0 + q; bt[q] = (t < Ntiles) ? t : 0; }

  const float* Ap = A + (long)(m0 + ar) * lda + ac;
  __bf16* d = As + ar * 32 + ac;

  v8f acc[4] = {};
  for (int kt = 0; kt < Ktiles; ++kt) {
    float4 f4 = *(const float4*)(Ap + (kt << 5));
    d[0] = f2bf(f4.x); d[1] = f2bf(f4.y); d[2] = f2bf(f4.z); d[3] = f2bf(f4.w);
    __syncthreads();
    v16bf a = load_frag(As, 32);
    const __bf16* bb = Bp + (long)kt * Ntiles * 512;
#pragma unroll
    for (int q = 0; q < 4; ++q) {
      v16bf b = load_frag(bb + (long)bt[q] * 512, 32);
      acc[q] = wmma_bf16(a, b, acc[q]);
    }
    __syncthreads();
  }

  // Epilogue.  C layout: lane l -> col (l&15), rows (l>>4)*8 + r
  const int mb = m0 + ((lane >> 4) << 3);
#pragma unroll
  for (int q = 0; q < 4; ++q) {
    int n = (nt0 + q) * 16 + (lane & 15);
    if (n < Ntot) {
      float bv = bias ? bias[n] : 0.0f;
#pragma unroll
      for (int r = 0; r < 8; ++r) {
        int m = mb + r;
        float v = acc[q][r] + bv;
        if (resid) v += resid[(long)m * ldr + n];
        if (act == 1)      v = tanhf(v);
        else if (act == 2) v = fmaxf(v, 0.0f);
        C[(long)m * ldc + n] = v;
      }
    }
  }
}

// ---------------------------------------------------------------------------
// M-major WMMA GEMM for skinny-N (Ntiles <= 4), e.g. the 294912-row route
// MLP.  Each wave owns its own M-tile and LDS quadrant: no barriers at all
// (per-wave DS ops are in order).  q-loop fully unrolled with uniform guards
// so acc[] stays statically indexed (no scratch spill).
// REQUIRES: M % 64 == 0; K % 4 == 0.
// ---------------------------------------------------------------------------
__global__ __launch_bounds__(128)
void gemm_mmajor_kernel(const float* __restrict__ A, int lda,
                        const __bf16* __restrict__ Bp, int Ntiles,
                        const float* __restrict__ bias,
                        const float* __restrict__ resid, int ldr,
                        float* __restrict__ C, int ldc,
                        int M, int Ntot, int K, int act) {
  __shared__ __align__(16) __bf16 As[4][512];

  const int tid  = threadIdx.x;
  const int wave = tid >> 5;
  const int lane = tid & 31;
  const int m0   = (blockIdx.x * 4 + wave) * 16;
  const int Ktiles = (K + 31) >> 5;

  v8f acc[4] = {};
  for (int kt = 0; kt < Ktiles; ++kt) {
    const int k0 = kt << 5;
    // stage this wave's A tile: 4 float4 groups per lane (K zero-padded)
#pragma unroll
    for (int q = 0; q < 4; ++q) {
      int e4 = lane + (q << 5);            // group 0..127
      int r = e4 >> 3, c = (e4 & 7) << 2;
      int gk = k0 + c;
      float4 f4 = make_float4(0.0f, 0.0f, 0.0f, 0.0f);
      if (gk < K) f4 = *(const float4*)(A + (long)(m0 + r) * lda + gk);
      __bf16* d = &As[wave][r * 32 + c];
      d[0] = f2bf(f4.x); d[1] = f2bf(f4.y); d[2] = f2bf(f4.z); d[3] = f2bf(f4.w);
    }
    v16bf a = load_frag(As[wave], 32);
    const __bf16* bb = Bp + (long)kt * Ntiles * 512;
#pragma unroll
    for (int q = 0; q < 4; ++q) {         // uniform guard keeps acc[] static
      if (q < Ntiles) {
        v16bf b = load_frag(bb + (long)q * 512, 32);
        acc[q] = wmma_bf16(a, b, acc[q]);
      }
    }
  }

  const int mb = m0 + ((lane >> 4) << 3);
#pragma unroll
  for (int q = 0; q < 4; ++q) {
    int n = q * 16 + (lane & 15);
    if (q < Ntiles && n < Ntot) {
      float bv = bias ? bias[n] : 0.0f;
#pragma unroll
      for (int r = 0; r < 8; ++r) {
        int m = mb + r;
        float v = acc[q][r] + bv;
        if (resid) v += resid[(long)m * ldr + n];
        if (act == 1)      v = tanhf(v);
        else if (act == 2) v = fmaxf(v, 0.0f);
        C[(long)m * ldc + n] = v;
      }
    }
  }
}

// ---------------------------------------------------------------------------
// Embedding gather + add:  out[t,c] = featH[t,c] + emb[ids[t], c]
// ---------------------------------------------------------------------------
__global__ void add_emb_kernel(const float* __restrict__ featH,
                               const float* __restrict__ emb,
                               const int* __restrict__ ids,
                               float* __restrict__ out) {
  int gid = blockIdx.x * 256 + threadIdx.x;     // 1536*256 total
  int t = gid >> 8, c = gid & 255;
  out[gid] = featH[gid] + emb[ids[t] * HID + c];
}

// ---------------------------------------------------------------------------
// LayerNorm over last dim 256 (one block of 256 threads per row)
// ---------------------------------------------------------------------------
__global__ __launch_bounds__(256)
void ln256_kernel(const float* __restrict__ X, const float* __restrict__ g,
                  const float* __restrict__ b, float* __restrict__ Y) {
  __shared__ float red[256];
  int row = blockIdx.x, t = threadIdx.x;
  float v = X[(long)row * 256 + t];
  red[t] = v; __syncthreads();
  for (int s = 128; s > 0; s >>= 1) { if (t < s) red[t] += red[t + s]; __syncthreads(); }
  float mean = red[0] * (1.0f / 256.0f);
  __syncthreads();
  float d = v - mean;
  red[t] = d * d; __syncthreads();
  for (int s = 128; s > 0; s >>= 1) { if (t < s) red[t] += red[t + s]; __syncthreads(); }
  float var = red[0] * (1.0f / 256.0f);
  float inv = rsqrtf(var + 1e-12f);
  Y[(long)row * 256 + t] = g[t] * d * inv + b[t];
}

// ---------------------------------------------------------------------------
// In-place LayerNorm over last dim 16 (one thread per row)
// ---------------------------------------------------------------------------
__global__ void ln16_kernel(float* __restrict__ X, const float* __restrict__ g,
                            const float* __restrict__ b, long rows) {
  long r = (long)blockIdx.x * blockDim.x + threadIdx.x;
  if (r >= rows) return;
  float x[16];
  float mu = 0.0f;
#pragma unroll
  for (int i = 0; i < 16; ++i) { x[i] = X[r * 16 + i]; mu += x[i]; }
  mu *= (1.0f / 16.0f);
  float s2 = 0.0f;
#pragma unroll
  for (int i = 0; i < 16; ++i) { float d = x[i] - mu; s2 += d * d; }
  s2 *= (1.0f / 16.0f);
  float inv = rsqrtf(s2 + 1e-12f);
#pragma unroll
  for (int i = 0; i < 16; ++i) X[r * 16 + i] = g[i] * (x[i] - mu) * inv + b[i];
}

// ---------------------------------------------------------------------------
// QrWd[t,h,r] = sum_k Qr[t,h,k] * Wd[r, h*16+k]   (rank-16 route-key fold)
// ---------------------------------------------------------------------------
__global__ __launch_bounds__(128)
void qrwd_kernel(const float* __restrict__ H2, const float* __restrict__ Wd_l,
                 float* __restrict__ QrWd) {
  int gid = blockIdx.x * 128 + threadIdx.x;   // 1536*128 total
  int t = gid >> 7;
  int h = (gid >> 4) & 7;
  int r = gid & 15;
  const float* hp = H2 + (long)t * (HEADS * TOT) + h * TOT + 2 * KS + VS; // Qr
  const float* wp = Wd_l + r * (HEADS * KRS) + h * KRS;
  float s = 0.0f;
#pragma unroll
  for (int k = 0; k < KRS; ++k) s += hp[k] * wp[k];
  QrWd[gid] = s;   // layout (t, h, r)
}

// ---------------------------------------------------------------------------
// Fused attention: one wave per (b, h, 16-row i-tile).
//   scores = SCALE*(Qn@Kn^T + <QrWd, R>) + mask ; softmax ; ctx = P@V + PR@Wroute_h
// ---------------------------------------------------------------------------
__global__ __launch_bounds__(32)
void attn_kernel(const float* __restrict__ H2, const float* __restrict__ R,
                 const float* __restrict__ QrWd, const float* __restrict__ amask,
                 const float* __restrict__ Wroute_l, float* __restrict__ ctx) {
  __shared__ __align__(16) __bf16 qn[16 * 32];
  __shared__ __align__(16) __bf16 knt[16 * 32];
  __shared__ __align__(16) __bf16 vt[32 * 192];   // V^T: [v][j]
  __shared__ __align__(16) __bf16 pb[16 * 192];   // probs bf16
  __shared__ float sc[16 * 192];                  // scores / probs f32
  __shared__ float qrw[16 * 16];
  __shared__ float pr[16 * 16];
  __shared__ float mval[192];

  const int it = blockIdx.x;        // 0..11
  const int bh = blockIdx.y;        // 0..63
  const int b = bh >> 3, h = bh & 7;
  const int lane = threadIdx.x;
  const int i0 = it * 16;
  const int LD2 = HEADS * TOT;      // 896

  // Stage Qn (16 x 32): one row per iter, lanes along k (coalesced)
#pragma unroll
  for (int r = 0; r < 16; ++r)
    qn[r * 32 + lane] = f2bf(H2[(long)(b * NN + i0 + r) * LD2 + h * TOT + lane]);
  // Stage V^T: vt[v*192 + j] = V[j, v]
  for (int v = 0; v < 32; ++v)
    for (int j = lane; j < 192; j += 32)
      vt[v * 192 + j] = f2bf(H2[(long)(b * NN + j) * LD2 + h * TOT + 2 * KS + v]);
  // Stage QrWd tile (16 rows x 16 r)
#pragma unroll
  for (int e = lane; e < 256; e += 32) {
    int r = e >> 4, rr = e & 15;
    qrw[e] = QrWd[(long)((b * NN + i0 + r) * HEADS + h) * KRS + rr];
  }
  // Mask values (key-dependent only)
  for (int j = lane; j < 192; j += 32)
    mval[j] = (1.0f - amask[b * NN + j]) * -10000.0f;

  // Scores: Qn @ Kn^T, one WMMA per 16-wide j-tile
  for (int jt = 0; jt < 12; ++jt) {
    __syncthreads();
#pragma unroll
    for (int jl = 0; jl < 16; ++jl)
      knt[jl * 32 + lane] =
          f2bf(H2[(long)(b * NN + jt * 16 + jl) * LD2 + h * TOT + KS + lane]);
    __syncthreads();
    v16bf a  = load_frag(qn, 32);
    v16bf kb = load_frag(knt, 32);
    v8f c = {};
    c = wmma_bf16(a, kb, c);
    int n = lane & 15, mb = (lane >> 4) << 3;
#pragma unroll
    for (int r = 0; r < 8; ++r) sc[(mb + r) * 192 + jt * 16 + n] = c[r];
  }
  __syncthreads();

  // Route-key bias (rank-16 contraction through R) + scale + mask
  for (int m = 0; m < 16; ++m) {
    const float* Rm = R + (long)(b * NN + i0 + m) * NN * RS;
    for (int j = lane; j < 192; j += 32) {
      const float* Rp = Rm + j * RS;
      float dotv = 0.0f;
#pragma unroll
      for (int r = 0; r < RS; ++r) dotv += qrw[m * 16 + r] * Rp[r];
      sc[m * 192 + j] = SCALE_F * (sc[m * 192 + j] + dotv) + mval[j];
    }
  }
  __syncthreads();

  // Row softmax: 2 lanes per row (96 cols each), combine via shfl_xor
  {
    int row = lane >> 1, j0 = (lane & 1) * 96;
    float mx = -1e30f;
    for (int j = j0; j < j0 + 96; ++j) mx = fmaxf(mx, sc[row * 192 + j]);
    mx = fmaxf(mx, __shfl_xor(mx, 1, 32));
    float s = 0.0f;
    for (int j = j0; j < j0 + 96; ++j) {
      float e2 = __expf(sc[row * 192 + j] - mx);
      sc[row * 192 + j] = e2; s += e2;
    }
    s += __shfl_xor(s, 1, 32);
    float inv = 1.0f / s;
    for (int j = j0; j < j0 + 96; ++j) {
      float p = sc[row * 192 + j] * inv;
      sc[row * 192 + j] = p;
      pb[row * 192 + j] = f2bf(p);
    }
  }
  __syncthreads();

  // PR[m,r] = sum_j p[m,j] * R[b,i0+m,j,r]   (rank-16 route-value fold)
  for (int e = lane; e < 256; e += 32) {
    int m = e >> 4, r = e & 15;
    const float* Rp = R + (long)(b * NN + i0 + m) * NN * RS + r;
    float s = 0.0f;
    for (int j = 0; j < 192; ++j) s += sc[m * 192 + j] * Rp[(long)j * RS];
    pr[e] = s;
  }
  __syncthreads();

  // ctx = P @ V  (WMMA over six 32-wide j-chunks, two 16-wide v-tiles)
  for (int vt0 = 0; vt0 < VS; vt0 += 16) {
    v8f c = {};
#pragma unroll
    for (int jc = 0; jc < 192; jc += 32) {
      v16bf a  = load_frag(pb + jc, 192);
      v16bf vb = load_frag(vt + (long)vt0 * 192 + jc, 192);
      c = wmma_bf16(a, vb, c);
    }
    int n = lane & 15, mb = (lane >> 4) << 3;
    int v = vt0 + n;
#pragma unroll
    for (int r8 = 0; r8 < 8; ++r8) {
      int m = mb + r8;
      float add = 0.0f;
#pragma unroll
      for (int rr = 0; rr < RS; ++rr)
        add += pr[m * 16 + rr] * Wroute_l[rr * (HEADS * VS) + h * VS + v];
      ctx[(long)(b * NN + i0 + m) * (HEADS * VS) + h * VS + v] = c[r8] + add;
    }
  }
}

// ---------------------------------------------------------------------------
// Host launchers
// ---------------------------------------------------------------------------
static inline int ntiles16(int n) { return (n + 15) / 16; }

static inline void launch_gemm(const float* A, int lda, const __bf16* Bp,
                               const float* bias, const float* resid, int ldr,
                               float* C, int ldc, int M, int N, int K, int act,
                               hipStream_t s) {
  int Nt = ntiles16(N);
  if (Nt <= 4) {
    dim3 grid(((M + 15) / 16 + 3) / 4);
    gemm_mmajor_kernel<<<grid, 128, 0, s>>>(A, lda, Bp, Nt, bias, resid, ldr,
                                            C, ldc, M, N, K, act);
  } else {
    dim3 grid((M + 15) / 16, (Nt + 15) / 16);
    gemm_nmajor_kernel<<<grid, 128, 0, s>>>(A, lda, Bp, Nt, bias, resid, ldr,
                                            C, ldc, M, N, K, act);
  }
}

static inline void launch_pack(const float* src, __bf16* dst, int K, int N,
                               hipStream_t s) {
  int Kt = (K + 31) / 32, Nt = ntiles16(N);
  long total = (long)Kt * Nt * 512;
  int blocks = (int)((total + 255) / 256);
  if (blocks > 8192) blocks = 8192;
  pack_bf16_kernel<<<blocks, 256, 0, s>>>(src, dst, K, N, Nt, total);
}

extern "C" void kernel_launch(void* const* d_in, const int* in_sizes, int n_in,
                              void* d_out, int out_size, void* d_ws, size_t ws_size,
                              hipStream_t stream) {
  const int*   node_ids      = (const int*)  d_in[0];
  const float* node_features = (const float*)d_in[1];
  const float* route_data    = (const float*)d_in[2];
  const float* amask         = (const float*)d_in[3];
  const float* emb_table     = (const float*)d_in[4];
  const float* fw1  = (const float*)d_in[5];
  const float* fb1  = (const float*)d_in[6];
  const float* fw2  = (const float*)d_in[7];
  const float* fb2  = (const float*)d_in[8];
  const float* eln_g = (const float*)d_in[9];
  const float* eln_b = (const float*)d_in[10];
  const float* rw1  = (const float*)d_in[11];
  const float* rb1  = (const float*)d_in[12];
  const float* rw2  = (const float*)d_in[13];
  const float* rb2  = (const float*)d_in[14];
  const float* rln_g = (const float*)d_in[15];
  const float* rln_b = (const float*)d_in[16];
  const float* W      = (const float*)d_in[17];
  const float* Wd     = (const float*)d_in[18];
  const float* Wroute = (const float*)d_in[19];
  const float* Wo     = (const float*)d_in[20];
  const float* bo     = (const float*)d_in[21];
  const float* oln_g  = (const float*)d_in[22];
  const float* oln_b  = (const float*)d_in[23];
  const float* Wi     = (const float*)d_in[24];
  const float* bi     = (const float*)d_in[25];
  const float* Wo2    = (const float*)d_in[26];
  const float* bo2    = (const float*)d_in[27];
  const float* fln_g  = (const float*)d_in[28];
  const float* fln_b  = (const float*)d_in[29];
  float* out = (float*)d_out;

  const int TOK = NB * NN;               // 1536
  const long RROWS = (long)NB * NN * NN; // 294912

  // ---- workspace layout (in floats) ----
  float* ws = (float*)d_ws;
  float* R1    = ws;                       // 4718592  (reused as H2 + inter)
  float* Rbuf  = ws + 4718592;             // 4718592
  float* feat1 = ws + 9437184;             // 393216   (reused as ctx)
  float* tmpb  = ws + 9830400;             // 393216   (featH, then LN temp)
  float* Hbuf  = ws + 10223616;            // 393216
  float* aout  = ws + 10616832;            // 393216
  float* qrwd  = ws + 11010048;            // 196608
  __bf16* bh   = (__bf16*)(ws + 11206656); // packed bf16 weight pool

  // packed tile sizes: Ktiles*Ntiles*512
  __bf16* fw1p  = bh;                //  1*16*512 =   8192
  __bf16* fw2p  = bh + 8192;         //  8*16*512 =  65536
  __bf16* rw1p  = bh + 73728;        //  1* 1*512 =    512
  __bf16* rw2p  = bh + 74240;        //  1* 1*512 =    512
  __bf16* Wp    = bh + 74752;        //  8*56*512 = 229376 per layer (x3)
  __bf16* Wop   = bh + 762880;       //  8*16*512 =  65536 per layer (x3)
  __bf16* Wip   = bh + 959488;       //  8*64*512 = 262144 per layer (x3)
  __bf16* Wo2p  = bh + 1745920;      // 32*16*512 = 262144 per layer (x3)
                                     // end: 2532352 bf16 = 1266176 floats

  float* H2    = R1;                 // 1376256 floats (R1 dead after LN of R)
  float* inter = R1 + 1376256;       // 1572864 floats
  float* ctx   = feat1;
  float* featH = tmpb;

  // ---- pack weights into WMMA fragment tiles (bf16) ----
  launch_pack(fw1, fw1p, NF, HID, stream);
  launch_pack(fw2, fw2p, HID, HID, stream);
  launch_pack(rw1, rw1p, RF, RS, stream);
  launch_pack(rw2, rw2p, RS, RS, stream);
  for (int l = 0; l < NL; ++l) {
    launch_pack(W   + (long)l * HID * 896,  Wp   + (long)l * 229376, HID, 896, stream);
    launch_pack(Wo  + (long)l * HID * HID,  Wop  + (long)l * 65536,  HID, HID, stream);
    launch_pack(Wi  + (long)l * HID * IM,   Wip  + (long)l * 262144, HID, IM,  stream);
    launch_pack(Wo2 + (long)l * IM  * HID,  Wo2p + (long)l * 262144, IM,  HID, stream);
  }

  // ---- embeddings ----
  launch_gemm(node_features, NF, fw1p, fb1, nullptr, 0,
              feat1, HID, TOK, HID, NF, /*tanh*/1, stream);
  launch_gemm(feat1, HID, fw2p, fb2, nullptr, 0,
              featH, HID, TOK, HID, HID, 0, stream);
  add_emb_kernel<<<TOK, 256, 0, stream>>>(featH, emb_table, node_ids, tmpb);
  ln256_kernel<<<TOK, 256, 0, stream>>>(tmpb, eln_g, eln_b, Hbuf);

  // ---- route embeddings: R = LN(tanh(route@rw1+rb1)@rw2+rb2) ----
  launch_gemm(route_data, RF, rw1p, rb1, nullptr, 0,
              R1, RS, (int)RROWS, RS, RF, /*tanh*/1, stream);
  launch_gemm(R1, RS, rw2p, rb2, nullptr, 0,
              Rbuf, RS, (int)RROWS, RS, RS, 0, stream);
  ln16_kernel<<<(int)((RROWS + 255) / 256), 256, 0, stream>>>(Rbuf, rln_g, rln_b, RROWS);
  // R1 region now dead -> reused for H2 / inter below.

  // ---- transformer layers ----
  for (int l = 0; l < NL; ++l) {
    // QKV(+Qr) projection: H2 = H @ W_l   [1536 x 896]
    launch_gemm(Hbuf, HID, Wp + (long)l * 229376, nullptr, nullptr, 0,
                H2, 896, TOK, 896, HID, 0, stream);
    // Fold Qr through Wd (rank-16 route-key term)
    qrwd_kernel<<<TOK, 128, 0, stream>>>(H2, Wd + (long)l * RS * (HEADS * KRS), qrwd);
    // Fused attention
    attn_kernel<<<dim3(NN / 16, NB * HEADS), 32, 0, stream>>>(
        H2, Rbuf, qrwd, amask, Wroute + (long)l * RS * (HEADS * VS), ctx);
    // Output projection + residual, then LN
    launch_gemm(ctx, HID, Wop + (long)l * 65536, bo + l * HID,
                Hbuf, HID, tmpb, HID, TOK, HID, HID, 0, stream);
    ln256_kernel<<<TOK, 256, 0, stream>>>(tmpb, oln_g + l * HID, oln_b + l * HID, aout);
    // FFN
    launch_gemm(aout, HID, Wip + (long)l * 262144, bi + l * IM,
                nullptr, 0, inter, IM, TOK, IM, HID, /*relu*/2, stream);
    launch_gemm(inter, IM, Wo2p + (long)l * 262144, bo2 + l * HID,
                aout, HID, tmpb, HID, TOK, HID, IM, 0, stream);
    ln256_kernel<<<TOK, 256, 0, stream>>>(tmpb, fln_g + l * HID, fln_b + l * HID,
                                          (l == NL - 1) ? out : Hbuf);
  }
}
